// CompressedInteractionNetwork_74646531604742
// MI455X (gfx1250) — matrix operand order repaired and checked by reading, not compiled, not run on hardware
//
#include <hip/hip_runtime.h>
#include <hip/hip_bf16.h>

typedef __attribute__((ext_vector_type(16))) _Float16 v16h;
typedef __attribute__((ext_vector_type(2)))  _Float16 h2;
typedef __attribute__((ext_vector_type(8)))  float    v8f;

#define BATCH   2048
#define NB      4         // batch elements per block (A-fragment reuse factor)
#define MDIM    40
#define DDIM    16
#define OCH     128
#define K0      1600      // 40*40
#define K12     2560      // 64*40
#define CHUNK   320       // = 8 h-rows * 40 m; divides 1600 and 2560
#define ZSTR    328       // padded row stride (halfs), rows 16B-aligned

// ---------------------------------------------------------------------------
// Prelude: pack fp32 weights into f16 in this code's per-lane 16-bit A-fragment
// order for v_wmma_f32_16x16x32_f16 (contiguous K, not AMD's table order; B matches):
//   lanes 0-15 : j=0..15 -> K=j      (K 0-15)
//   lanes 16-31: j=0..15 -> K=j+16   (K 16-31)
// Flat index: (((wave*KC + kc)*32 + lane)*16 + j)
// ---------------------------------------------------------------------------
__global__ __launch_bounds__(256) void pack_w_kernel(const float* __restrict__ w,
                                                     _Float16* __restrict__ pw,
                                                     int K, int total) {
    int idx = blockIdx.x * blockDim.x + threadIdx.x;
    if (idx >= total) return;
    int j    = idx & 15;
    int lane = (idx >> 4) & 31;
    int rest = idx >> 9;
    int KC   = K >> 5;
    int kc   = rest % KC;
    int wave = rest / KC;
    int o    = wave * 16 + (lane & 15);
    int k    = kc * 32 + j + ((lane >> 4) << 4);
    pw[idx]  = (_Float16)w[o * K + k];
}

// ---------------------------------------------------------------------------
// Fused CIN kernel: one block = NB batch elements, 8 waves, 3 layers fused.
// Each A-fragment (packed weights) load feeds NB WMMAs -> 4x less L2 traffic.
// ---------------------------------------------------------------------------
__global__ __launch_bounds__(256) void cin_kernel(
    const float*    __restrict__ x,
    const _Float16* __restrict__ pw0,
    const _Float16* __restrict__ pw1,
    const _Float16* __restrict__ pw2,
    const float*    __restrict__ b0,
    const float*    __restrict__ b1,
    const float*    __restrict__ b2,
    const float*    __restrict__ wl,
    float*          __restrict__ outp)
{
    __shared__ float    xf[NB][MDIM * DDIM];       // x                (10 KB)
    __shared__ float    prevf[NB][64 * DDIM];      // prev activations (16 KB)
    __shared__ float    outb[NB][OCH * DDIM];      // layer outputs    (32 KB)
    __shared__ _Float16 zt[2][NB][16 * ZSTR];      // Z^T double buf   (82 KB)
    __shared__ float    red[NB][64];               // feat*wl partials ( 1 KB)

    const int t    = threadIdx.x;
    const int b0i  = blockIdx.x * NB;
    const int lane = t & 31;
    const int wave = t >> 5;

    // ---- load x for NB batches into LDS; prev(layer0) = x; zero red ----
    for (int i = t; i < NB * MDIM * DDIM; i += 256) {
        int nb = i / (MDIM * DDIM), ii = i % (MDIM * DDIM);
        float v = x[(size_t)(b0i + nb) * MDIM * DDIM + ii];
        xf[nb][ii]    = v;
        prevf[nb][ii] = v;
    }
    red[t >> 6][t & 63] = 0.0f;      // 256 threads cover all NB*64 entries
    __syncthreads();

    // ---- static Z-build mapping: 256 thr = NB batches x (16 d x 4 groups) ----
    // group g: m-half (g&1)*20, h-quad (g>>1)*4. Preload 20 x[m,d] values as
    // 10 packed f16 pairs, reused for every chunk of every layer.
    const int znb = t >> 6;            // batch this thread builds
    const int zd  = t & 15;            // d column
    const int g   = (t >> 4) & 3;
    const int mb  = (g & 1) * 20;      // m range base
    const int hs  = (g >> 1) * 4;      // h sub-offset within chunk (0 or 4)
    h2 xh2[10];
    #pragma unroll
    for (int i = 0; i < 10; ++i) {
        int m = mb + 2 * i;
        xh2[i] = h2{ (_Float16)xf[znb][m * DDIM + zd],
                     (_Float16)xf[znb][(m + 1) * DDIM + zd] };
    }

    // ---- consume-side fragment addressing (constant across layers) ----
    const int dcol  = lane & 15;
    const int khalf = (lane >> 4) << 4;   // B-frag: lanes 16-31 take K+16

    #pragma unroll
    for (int layer = 0; layer < 3; ++layer) {
        const int       K    = (layer == 0) ? K0 : K12;
        const int       KC   = K >> 5;
        const int       NCH  = K / CHUNK;                // 5 or 8
        const _Float16* pw   = (layer == 0) ? pw0 : ((layer == 1) ? pw1 : pw2);
        const float*    bias = (layer == 0) ? b0  : ((layer == 1) ? b1  : b2);

        v8f c[NB];
        #pragma unroll
        for (int nb = 0; nb < NB; ++nb) c[nb] = v8f{};

        // ---- build chunk 0 into buffer 0 ----
        #pragma unroll
        for (int hh = 0; hh < 4; ++hh) {
            h2 pp; pp[0] = pp[1] = (_Float16)prevf[znb][(hs + hh) * DDIM + zd];
            _Float16* zr = &zt[0][znb][zd * ZSTR + (hs + hh) * 40 + mb];
            #pragma unroll
            for (int i = 0; i < 10; ++i) *(h2*)(zr + 2 * i) = pp * xh2[i];
        }
        __syncthreads();

        for (int cidx = 0; cidx < NCH; ++cidx) {
            const int bufc = cidx & 1;
            const int kbase = cidx * (CHUNK / 32);
            // ---- consume: per kc one A load reused by NB WMMAs ----
            #pragma unroll
            for (int kc = 0; kc < CHUNK / 32; ++kc) {
                union { v16h v; uint4 u[2]; } A;
                A.v = *(const v16h*)(pw + (((size_t)wave * KC + kbase + kc) * 32
                                           + lane) * 16);
                #pragma unroll
                for (int nb = 0; nb < NB; ++nb) {
                    union { v16h v; uint4 u[2]; } Bm;
                    const uint4* zp = (const uint4*)
                        (&zt[bufc][nb][dcol * ZSTR + khalf + kc * 32]);
                    Bm.u[0] = zp[0];
                    Bm.u[1] = zp[1];
                    c[nb] = __builtin_amdgcn_wmma_f32_16x16x32_f16(
                                false, A.v, false, Bm.v, (short)0, c[nb],
                                false, false);
                }
            }
            // ---- build next chunk into the other buffer (overlaps consume) ----
            if (cidx + 1 < NCH) {
                const int h0 = (cidx + 1) * 8;
                #pragma unroll
                for (int hh = 0; hh < 4; ++hh) {
                    h2 pp; pp[0] = pp[1] =
                        (_Float16)prevf[znb][(h0 + hs + hh) * DDIM + zd];
                    _Float16* zr = &zt[bufc ^ 1][znb]
                                      [zd * ZSTR + (hs + hh) * 40 + mb];
                    #pragma unroll
                    for (int i = 0; i < 10; ++i) *(h2*)(zr + 2 * i) = pp * xh2[i];
                }
            }
            __syncthreads();
        }

        // ---- epilogue: bias + relu, scatter C tiles (M = r + 8*(lane>=16)) ----
        const int mo = wave * 16 + ((lane >> 4) << 3);
        #pragma unroll
        for (int r = 0; r < 8; ++r) {
            int o = mo + r;
            float bo = bias[o];
            #pragma unroll
            for (int nb = 0; nb < NB; ++nb) {
                float v = c[nb][r] + bo;
                outb[nb][o * DDIM + dcol] = fmaxf(v, 0.0f);
            }
        }
        __syncthreads();
        // rows 64..127 -> feat*wl partials; rows 0..63 -> prev
        {
            int nb = t >> 6, tt = t & 63;
            float s = 0.0f;
            #pragma unroll
            for (int dd = 0; dd < DDIM; ++dd) s += outb[nb][(64 + tt) * DDIM + dd];
            red[nb][tt] += s * wl[layer * 64 + tt];
        }
        for (int i = t; i < NB * 64 * DDIM; i += 256) {
            int nb = i >> 10, ii = i & 1023;
            prevf[nb][ii] = outb[nb][ii];
        }
        __syncthreads();
    }

    // ---- final: one scalar per batch element ----
    if (t < NB) {
        float s = 0.0f;
        #pragma unroll
        for (int i = 0; i < 64; ++i) s += red[t][i];
        outp[b0i + t] = s;
    }
}

// ---------------------------------------------------------------------------
extern "C" void kernel_launch(void* const* d_in, const int* in_sizes, int n_in,
                              void* d_out, int out_size, void* d_ws, size_t ws_size,
                              hipStream_t stream) {
    const float* x  = (const float*)d_in[0];
    const float* w0 = (const float*)d_in[1];
    const float* b0 = (const float*)d_in[2];
    const float* w1 = (const float*)d_in[3];
    const float* b1 = (const float*)d_in[4];
    const float* w2 = (const float*)d_in[5];
    const float* b2 = (const float*)d_in[6];
    const float* wl = (const float*)d_in[7];

    _Float16* pw0 = (_Float16*)d_ws;
    _Float16* pw1 = pw0 + (size_t)OCH * K0;    // 32B-aligned offsets
    _Float16* pw2 = pw1 + (size_t)OCH * K12;

    const int tot0 = OCH * K0;
    const int tot1 = OCH * K12;
    pack_w_kernel<<<(tot0 + 255) / 256, 256, 0, stream>>>(w0, pw0, K0,  tot0);
    pack_w_kernel<<<(tot1 + 255) / 256, 256, 0, stream>>>(w1, pw1, K12, tot1);
    pack_w_kernel<<<(tot1 + 255) / 256, 256, 0, stream>>>(w2, pw2, K12, tot1);

    cin_kernel<<<BATCH / NB, 256, 0, stream>>>(x, pw0, pw1, pw2, b0, b1, b2, wl,
                                               (float*)d_out);
}
